// CoupledMambaLayer_17239998726199
// MI455X (gfx1250) — matrix operand order repaired and checked by reading, not compiled, not run on hardware
//
#include <hip/hip_runtime.h>
#include <hip/hip_bf16.h>

#define D_MODEL 512
#define D_STATE 16
#define D_INNER 1024
#define DT_RANK 32
#define SEQ_B   2
#define SEQ_L   2048
#define ROWS    (SEQ_B * SEQ_L)   // 4096

typedef __attribute__((ext_vector_type(16))) __bf16 v16bf;
typedef __attribute__((ext_vector_type(8)))  __bf16 v8bf;
typedef __attribute__((ext_vector_type(8)))  float  v8f;

__device__ __forceinline__ __bf16 f2bf(float f) {
    union { float f; unsigned u; } v; v.f = f;
    unsigned r = v.u + 0x7FFFu + ((v.u >> 16) & 1u);   // round-to-nearest-even
    unsigned short h = (unsigned short)(r >> 16);
    __bf16 out;
    __builtin_memcpy(&out, &h, 2);
    return out;
}

// A-fragment (16x32 bf16, ISA layout): elems 0..7 = K[half*8 .. +7],
// elems 8..15 = K[16+half*8 .. +7]; p already points at row + half*8.
__device__ __forceinline__ v16bf load_afrag(const __bf16* p) {
    v8bf lo = *(const v8bf*)(p);
    v8bf hi = *(const v8bf*)(p + 16);
    v16bf r;
#pragma unroll
    for (int i = 0; i < 8; ++i) { r[i] = lo[i]; r[8 + i] = hi[i]; }
    return r;
}

// ---------------------------------------------------------------------------
// f32 -> bf16 conversion
// ---------------------------------------------------------------------------
__global__ __launch_bounds__(256) void cvt_f32_bf16(const float* __restrict__ in,
                                                    __bf16* __restrict__ out, int n) {
    int i = blockIdx.x * 256 + threadIdx.x;
    if (i < n) out[i] = f2bf(in[i]);
}

// ---------------------------------------------------------------------------
// WMMA bf16 GEMM:  out[M,N] = A[M,K] @ W[N,K]^T  (row-major bf16), f32 acc
// Each wave computes a 64x32 output block as 4x2 WMMA tiles: 12 b128 loads
// feed 8 v_wmma_f32_16x16x32_bf16 per K-step (1.5 loads/WMMA).
// mode 0: out = acc
// mode 1: out = resid + coef[coefIdx] * acc          (cross-modal couple)
// mode 2: out = softplus(acc + bias[n])              (delta projection)
// mode 3: out = resid + acc                          (out_proj residual)
// ---------------------------------------------------------------------------
__global__ __launch_bounds__(256)
void gemm_wmma_bf16(const __bf16* __restrict__ A, const __bf16* __restrict__ W,
                    float* __restrict__ out, int M, int N, int K,
                    int mode, const float* __restrict__ resid,
                    const float* __restrict__ coefp, int coefIdx,
                    const float* __restrict__ bias)
{
    const int lane = threadIdx.x & 31;
    const int wave = threadIdx.x >> 5;
    const int tN   = N >> 5;                      // 32-wide wave tiles
    const int wid  = blockIdx.x * 8 + wave;
    if (wid >= (M >> 6) * tN) return;             // wave-uniform
    const int wm0 = (wid / tN) << 6;              // 64 rows per wave
    const int wn0 = (wid % tN) << 5;              // 32 cols per wave

    const int half = lane >> 4;                   // 0 or 1
    const int lid  = lane & 15;

    const __bf16* Ar[4];
    Ar[0] = A + (size_t)(wm0 + lid) * K + half * 8;
#pragma unroll
    for (int i = 1; i < 4; ++i) Ar[i] = Ar[i - 1] + (size_t)16 * K;
    const __bf16* Wr0 = W + (size_t)(wn0 + lid) * K + half * 8;
    const __bf16* Wr1 = Wr0 + (size_t)16 * K;

    v8f acc[4][2];
#pragma unroll
    for (int i = 0; i < 4; ++i)
#pragma unroll
        for (int j = 0; j < 2; ++j) acc[i][j] = {};

    for (int k0 = 0; k0 < K; k0 += 32) {
        v16bf af[4], bfm[2];
#pragma unroll
        for (int i = 0; i < 4; ++i) af[i] = load_afrag(Ar[i] + k0);
        bfm[0] = load_afrag(Wr0 + k0);
        bfm[1] = load_afrag(Wr1 + k0);
        if (k0 + 256 < K) {                       // prefetch 8 K-steps ahead
#pragma unroll
            for (int i = 0; i < 4; ++i) __builtin_prefetch(Ar[i] + k0 + 256, 0, 3);
            __builtin_prefetch(Wr0 + k0 + 256, 0, 3);
            __builtin_prefetch(Wr1 + k0 + 256, 0, 3);
        }
#pragma unroll
        for (int i = 0; i < 4; ++i)
#pragma unroll
            for (int j = 0; j < 2; ++j)
                acc[i][j] = __builtin_amdgcn_wmma_f32_16x16x32_bf16(
                    false, af[i], false, bfm[j], (short)0, acc[i][j], false, false);
    }

    const float cf = (mode == 1) ? coefp[coefIdx] : 0.f;
#pragma unroll
    for (int j = 0; j < 2; ++j) {
        const int n  = wn0 + j * 16 + lid;
        const float bn = (mode == 2) ? bias[n] : 0.f;
#pragma unroll
        for (int i = 0; i < 4; ++i) {
#pragma unroll
            for (int r = 0; r < 8; ++r) {
                const int mm = wm0 + i * 16 + half * 8 + r;
                const size_t o = (size_t)mm * N + n;
                float v = acc[i][j][r];
                if (mode == 1)      out[o] = resid[o] + cf * v;
                else if (mode == 2) { float s = v + bn; out[o] = (s > 20.f) ? s : log1pf(__expf(s)); }
                else if (mode == 3) out[o] = resid[o] + v;
                else                out[o] = v;
            }
        }
    }
}

// ---------------------------------------------------------------------------
// LayerNorm over D_MODEL=512, one row per wave (wave32), bf16 output
// ---------------------------------------------------------------------------
__global__ __launch_bounds__(256)
void layernorm_bf16(const float* __restrict__ x, const float* __restrict__ nw,
                    const float* __restrict__ nb, __bf16* __restrict__ out, int rows)
{
    const int lane = threadIdx.x & 31;
    const int wave = threadIdx.x >> 5;
    const int row  = blockIdx.x * 8 + wave;
    if (row >= rows) return;
    const float* xr = x + (size_t)row * D_MODEL;
    float vals[16], s = 0.f, s2 = 0.f;
#pragma unroll
    for (int i = 0; i < 16; ++i) {
        float v = xr[lane * 16 + i];
        vals[i] = v; s += v; s2 += v * v;
    }
#pragma unroll
    for (int m2 = 16; m2 >= 1; m2 >>= 1) {
        s  += __shfl_xor(s,  m2, 32);
        s2 += __shfl_xor(s2, m2, 32);
    }
    const float mu  = s * (1.f / D_MODEL);
    const float var = s2 * (1.f / D_MODEL) - mu * mu;
    const float rs  = rsqrtf(var + 1e-5f);
    __bf16* orow = out + (size_t)row * D_MODEL;
#pragma unroll
    for (int i = 0; i < 16; ++i) {
        int c = lane * 16 + i;
        orow[c] = f2bf((vals[i] - mu) * rs * nw[c] + nb[c]);
    }
}

// ---------------------------------------------------------------------------
// Depthwise causal conv (D_CONV=4) + bias + SiLU.  Reads x from first half of
// xz [ROWS, 2*D_INNER]; writes f32 activation, bf16 copy, and extracts z gate.
// ---------------------------------------------------------------------------
__global__ __launch_bounds__(256)
void conv_silu(const float* __restrict__ xz, const float* __restrict__ cw,
               const float* __restrict__ cb, float* __restrict__ xc_act,
               __bf16* __restrict__ xc_bf, float* __restrict__ zbuf)
{
    const int idx = blockIdx.x * 256 + threadIdx.x;     // over ROWS*D_INNER
    const int d  = idx & (D_INNER - 1);
    const int bl = idx >> 10;
    const int l  = bl & (SEQ_L - 1);
    float acc = cb[d];
#pragma unroll
    for (int j = 0; j < 4; ++j) {
        int ll = l - 3 + j;
        if (ll >= 0) acc += cw[d * 4 + j] * xz[(size_t)(bl - 3 + j) * (2 * D_INNER) + d];
    }
    float s = acc / (1.f + __expf(-acc));
    xc_act[idx] = s;
    xc_bf[idx]  = f2bf(s);
    zbuf[idx]   = xz[(size_t)bl * (2 * D_INNER) + D_INNER + d];
}

// ---------------------------------------------------------------------------
// Extract dt columns (0..31) of dbc [ROWS,64] as bf16 [ROWS,32]
// ---------------------------------------------------------------------------
__global__ __launch_bounds__(256)
void extract_dt(const float* __restrict__ dbc, __bf16* __restrict__ dtbf) {
    const int i = blockIdx.x * 256 + threadIdx.x;       // ROWS*DT_RANK
    const int r = i >> 5, j = i & 31;
    dtbf[i] = f2bf(dbc[r * 64 + j]);
}

// ---------------------------------------------------------------------------
// Selective scan, ALL modalities in one launch: grid = (D_INNER/256, B, 3).
// Thread per channel d, 16-state recurrence in registers.  Fuses
// y = (scan + u*D_skip) * silu(z) and bf16 conversion for out_proj.
// ---------------------------------------------------------------------------
__global__ __launch_bounds__(256)
void ssm_scan_all(const float* __restrict__ delta3, const float* __restrict__ z3,
                  const float* __restrict__ xc3, const float* __restrict__ dbc3,
                  const float* __restrict__ A_log, const float* __restrict__ D_skip,
                  __bf16* __restrict__ ybf3)
{
    const int d = blockIdx.x * 256 + threadIdx.x;
    const int b = blockIdx.y;
    const int m = blockIdx.z;
    const float* delta = delta3 + (size_t)m * ROWS * D_INNER;
    const float* zb    = z3     + (size_t)m * ROWS * D_INNER;
    const float* xc    = xc3    + (size_t)m * ROWS * D_INNER;
    const float* dbc   = dbc3   + (size_t)m * ROWS * 64;
    __bf16*      ybf   = ybf3   + (size_t)m * ROWS * D_INNER;
    const float* Alg   = A_log  + (size_t)m * D_INNER * D_STATE;

    float Aneg[D_STATE], h[D_STATE];
#pragma unroll
    for (int n = 0; n < D_STATE; ++n) { Aneg[n] = -__expf(Alg[d * D_STATE + n]); h[n] = 0.f; }
    const float Dsk = D_skip[m * D_INNER + d];
    for (int l = 0; l < SEQ_L; ++l) {
        const size_t row = (size_t)b * SEQ_L + l;
        const float dt = delta[row * D_INNER + d];
        const float u  = xc[row * D_INNER + d];
        const float du = dt * u;
        const float* bc = dbc + row * 64;
        float y = 0.f;
#pragma unroll
        for (int n = 0; n < D_STATE; ++n) {
            h[n] = __expf(dt * Aneg[n]) * h[n] + du * bc[DT_RANK + n];
            y += h[n] * bc[DT_RANK + D_STATE + n];
        }
        const float z  = zb[row * D_INNER + d];
        const float yo = (y + u * Dsk) * (z / (1.f + __expf(-z)));
        ybf[row * D_INNER + d] = f2bf(yo);
    }
}

// ---------------------------------------------------------------------------
extern "C" void kernel_launch(void* const* d_in, const int* in_sizes, int n_in,
                              void* d_out, int out_size, void* d_ws, size_t ws_size,
                              hipStream_t stream) {
    const float* v       = (const float*)d_in[0];
    const float* a       = (const float*)d_in[1];
    const float* t       = (const float*)d_in[2];
    const float* norm_w  = (const float*)d_in[3];
    const float* norm_b  = (const float*)d_in[4];
    const float* in_w    = (const float*)d_in[5];
    const float* conv_w  = (const float*)d_in[6];
    const float* conv_b  = (const float*)d_in[7];
    const float* xproj_w = (const float*)d_in[8];
    const float* dt_w    = (const float*)d_in[9];
    const float* dt_b    = (const float*)d_in[10];
    const float* A_log   = (const float*)d_in[11];
    const float* D_skip  = (const float*)d_in[12];
    const float* out_w   = (const float*)d_in[13];
    const float* cpl_w   = (const float*)d_in[14];
    const float* coef    = (const float*)d_in[15];
    float* out = (float*)d_out;

    char* ws = (char*)d_ws;
    size_t off = 0;
    auto alloc = [&](size_t bytes) -> void* {
        void* p = ws + off;
        off += (bytes + 255) & ~(size_t)255;
        return p;
    };

    // bf16 weight copies (all 3 modalities)
    __bf16* cw_bf  = (__bf16*)alloc((size_t)3 * 512 * 512 * 2);
    __bf16* inw_bf = (__bf16*)alloc((size_t)3 * 2048 * 512 * 2);
    __bf16* xp_bf  = (__bf16*)alloc((size_t)3 * 64 * 1024 * 2);
    __bf16* dtw_bf = (__bf16*)alloc((size_t)3 * 1024 * 32 * 2);
    __bf16* ow_bf  = (__bf16*)alloc((size_t)3 * 512 * 1024 * 2);
    __bf16* vbf    = (__bf16*)alloc((size_t)ROWS * D_MODEL * 2);
    __bf16* abf    = (__bf16*)alloc((size_t)ROWS * D_MODEL * 2);
    __bf16* tbf    = (__bf16*)alloc((size_t)ROWS * D_MODEL * 2);
    // per-modality persistent buffers (needed concurrently for fused scan)
    float*  x_in3  = (float*) alloc((size_t)3 * ROWS * D_MODEL * 4);
    float*  xc3    = (float*) alloc((size_t)3 * ROWS * D_INNER * 4);
    float*  z3     = (float*) alloc((size_t)3 * ROWS * D_INNER * 4);
    float*  dbc3   = (float*) alloc((size_t)3 * ROWS * 64 * 4);
    float*  delta3 = (float*) alloc((size_t)3 * ROWS * D_INNER * 4);
    __bf16* ybf3   = (__bf16*)alloc((size_t)3 * ROWS * D_INNER * 2);
    // shared scratch (reused per modality before the scan)
    __bf16* xn_bf  = (__bf16*)alloc((size_t)ROWS * D_MODEL * 2);
    float*  xz     = (float*) alloc((size_t)ROWS * 2 * D_INNER * 4);
    __bf16* xc_bf  = (__bf16*)alloc((size_t)ROWS * D_INNER * 2);
    __bf16* dt_bf  = (__bf16*)alloc((size_t)ROWS * DT_RANK * 2);

    auto cvt = [&](const float* src, __bf16* dst, int n) {
        cvt_f32_bf16<<<(n + 255) / 256, 256, 0, stream>>>(src, dst, n);
    };
    cvt(cpl_w,   cw_bf,  3 * 512 * 512);
    cvt(in_w,    inw_bf, 3 * 2048 * 512);
    cvt(xproj_w, xp_bf,  3 * 64 * 1024);
    cvt(dt_w,    dtw_bf, 3 * 1024 * 32);
    cvt(out_w,   ow_bf,  3 * 512 * 1024);
    cvt(v, vbf, ROWS * D_MODEL);
    cvt(a, abf, ROWS * D_MODEL);
    cvt(t, tbf, ROWS * D_MODEL);

    const float*  bases[3] = { v, a, t };
    __bf16*       srcs[3]  = { abf, tbf, vbf };   // couple source per modality

    auto gemm = [&](const __bf16* A_, const __bf16* W_, float* O_, int M, int N, int K,
                    int mode, const float* resid, int coefIdx, const float* bias) {
        int waves  = (M / 64) * (N / 32);         // one 64x32 block per wave
        int blocks = (waves + 7) / 8;
        gemm_wmma_bf16<<<blocks, 256, 0, stream>>>(A_, W_, O_, M, N, K,
                                                   mode, resid, coef, coefIdx, bias);
    };

    // ---- Phase A: everything up to the scan, per modality -----------------
    for (int m = 0; m < 3; ++m) {
        float* x_in  = x_in3  + (size_t)m * ROWS * D_MODEL;
        float* xcact = xc3    + (size_t)m * ROWS * D_INNER;
        float* zbuf  = z3     + (size_t)m * ROWS * D_INNER;
        float* dbc   = dbc3   + (size_t)m * ROWS * 64;
        float* dlt   = delta3 + (size_t)m * ROWS * D_INNER;

        // 1) cross-modal couple: x_in = base + coef[m] * (src @ couple_w[m]^T)
        gemm(srcs[m], cw_bf + (size_t)m * 512 * 512, x_in,
             ROWS, D_MODEL, D_MODEL, 1, bases[m], m, nullptr);
        // 2) layernorm -> bf16
        layernorm_bf16<<<ROWS / 8, 256, 0, stream>>>(x_in, norm_w + m * D_MODEL,
                                                     norm_b + m * D_MODEL, xn_bf, ROWS);
        // 3) in_proj: xz = xn @ in_w^T
        gemm(xn_bf, inw_bf + (size_t)m * 2048 * 512, xz,
             ROWS, 2 * D_INNER, D_MODEL, 0, nullptr, 0, nullptr);
        // 4) depthwise causal conv + SiLU, extract z gate
        conv_silu<<<(ROWS * D_INNER) / 256, 256, 0, stream>>>(
            xz, conv_w + (size_t)m * D_INNER * 4, conv_b + m * D_INNER,
            xcact, xc_bf, zbuf);
        // 5) x_proj: dbc = xc @ xproj_w^T
        gemm(xc_bf, xp_bf + (size_t)m * 64 * 1024, dbc,
             ROWS, 64, D_INNER, 0, nullptr, 0, nullptr);
        // 6) dt slice -> bf16
        extract_dt<<<(ROWS * DT_RANK) / 256, 256, 0, stream>>>(dbc, dt_bf);
        // 7) delta = softplus(dt @ dt_w^T + dt_b)
        gemm(dt_bf, dtw_bf + (size_t)m * 1024 * 32, dlt,
             ROWS, D_INNER, DT_RANK, 2, nullptr, 0, dt_b + m * D_INNER);
    }

    // ---- Phase B: all three scans concurrently (3x parallelism on the
    //      latency-bound stage) --------------------------------------------
    ssm_scan_all<<<dim3(D_INNER / 256, SEQ_B, 3), 256, 0, stream>>>(
        delta3, z3, xc3, dbc3, A_log, D_skip, ybf3);

    // ---- Phase C: out_proj + residual ------------------------------------
    for (int m = 0; m < 3; ++m) {
        gemm(ybf3 + (size_t)m * ROWS * D_INNER, ow_bf + (size_t)m * 512 * 1024,
             out + (size_t)m * ROWS * D_MODEL, ROWS, D_MODEL, D_INNER,
             3, x_in3 + (size_t)m * ROWS * D_MODEL, 0, nullptr);
    }
}
